// Network_12627203850256
// MI455X (gfx1250) — matrix operand order repaired and checked by reading, not compiled, not run on hardware
//
#include <hip/hip_runtime.h>

typedef __attribute__((ext_vector_type(16))) _Float16 v16h;
typedef __attribute__((ext_vector_type(8)))  _Float16 v8h;
typedef __attribute__((ext_vector_type(8)))  float    v8f;

#define N1CV   400000
#define N2CV   110592
#define BN_EPS 1e-5f

// ---- pack W2 [27][64][128] f32 -> WMMA B-fragment f16 layout -------------
// pw2[((k*8+nb)*2+kc)*32 + lane][i] = W2[k][kc*32 + (i&7) + (i&8 ? 16 : 0) + (lane&16 ? 8 : 0)][nb*16 + (lane&15)]
__global__ void __launch_bounds__(256) pack_w2_kernel(const float* __restrict__ W2,
                                                      _Float16* __restrict__ pw2) {
  int idx = blockIdx.x * 256 + threadIdx.x;   // 27*8*2*32*16 = 221184 threads
  int i  = idx & 15;
  int L  = (idx >> 4) & 31;
  int kc = (idx >> 9) & 1;
  int nb = (idx >> 10) & 7;
  int k  = idx >> 13;
  int K  = kc * 32 + (i & 7) + ((i & 8) << 1) + ((L & 16) >> 1);
  int n  = nb * 16 + (L & 15);
  pw2[idx] = (_Float16)W2[((size_t)k * 64 + K) * 128 + n];
}

// ---- conv1: one offset k, scalar f32, race-free RMW scatter --------------
__global__ void __launch_bounds__(256) conv1_pass(const float* __restrict__ feats,
                                                  const float* __restrict__ W1,
                                                  const int* __restrict__ im,
                                                  const int* __restrict__ om,
                                                  float* __restrict__ out1, int k) {
  int idx = blockIdx.x * 256 + threadIdx.x;   // N1CV*4 threads
  int e  = idx >> 2;
  int cb = (idx & 3) * 16;
  int row  = im[(size_t)k * N1CV + e];
  int orow = om[(size_t)k * N1CV + e];
  const float* xr = feats + (size_t)row * 3;
  float x0 = xr[0], x1 = xr[1], x2 = xr[2];
  const float* w = W1 + (size_t)k * 192;
  float* o = out1 + (size_t)orow * 64;
#pragma unroll
  for (int j = 0; j < 4; ++j) {
    int c = cb + j * 4;
    float4 wa = *(const float4*)(w + c);
    float4 wb = *(const float4*)(w + 64 + c);
    float4 wc = *(const float4*)(w + 128 + c);
    float4* po = (float4*)(o + c);
    float4 acc = *po;
    acc.x += x0 * wa.x + x1 * wb.x + x2 * wc.x;
    acc.y += x0 * wa.y + x1 * wb.y + x2 * wc.y;
    acc.z += x0 * wa.z + x1 * wb.z + x2 * wc.z;
    acc.w += x0 * wa.w + x1 * wb.w + x2 * wc.w;
    *po = acc;
  }
}

// ---- helpers for conv2 ----------------------------------------------------
__device__ __forceinline__ v16h load_a_frag(const _Float16* __restrict__ arow, int kc) {
  v8h lo = *(const v8h*)(arow + kc * 32);        // lanes<16: K 0-7   | lanes>=16: K 8-15
  v8h hh = *(const v8h*)(arow + kc * 32 + 16);   // lanes<16: K 16-23 | lanes>=16: K 24-31
  v16h a;
#pragma unroll
  for (int i = 0; i < 8; ++i) { a[i] = lo[i]; a[8 + i] = hh[i]; }
  return a;
}

// batched read-modify-write scatter: 8 gathers clause, then 8 stores
__device__ __forceinline__ void scatter_add(float* __restrict__ out2,
                                            const int* __restrict__ omk,
                                            int hi, int n, v8f c) {
  int4 oa = *(const int4*)(omk + hi * 8);
  int4 ob = *(const int4*)(omk + hi * 8 + 4);
  float* p0 = out2 + (size_t)oa.x * 128 + n;
  float* p1 = out2 + (size_t)oa.y * 128 + n;
  float* p2 = out2 + (size_t)oa.z * 128 + n;
  float* p3 = out2 + (size_t)oa.w * 128 + n;
  float* p4 = out2 + (size_t)ob.x * 128 + n;
  float* p5 = out2 + (size_t)ob.y * 128 + n;
  float* p6 = out2 + (size_t)ob.z * 128 + n;
  float* p7 = out2 + (size_t)ob.w * 128 + n;
  float v0 = *p0, v1 = *p1, v2 = *p2, v3 = *p3;
  float v4 = *p4, v5 = *p5, v6 = *p6, v7 = *p7;
  *p0 = v0 + c[0]; *p1 = v1 + c[1]; *p2 = v2 + c[2]; *p3 = v3 + c[3];
  *p4 = v4 + c[4]; *p5 = v5 + c[5]; *p6 = v6 + c[6]; *p7 = v7 + c[7];
}

// ---- conv2: one offset k, f16 WMMA 16x16x32, f32 accumulate --------------
// block = 256 (8 waves); wave nb owns 16 output channels.
// Each block handles 4 tiles (16 entries each); B hoisted; tiles processed in
// pairs with two independent accumulator chains to hide WMMA hazards.
__global__ void __launch_bounds__(256) conv2_wmma(const _Float16* __restrict__ o1h,
                                                  const _Float16* __restrict__ pw2,
                                                  const int* __restrict__ im,
                                                  const int* __restrict__ om,
                                                  float* __restrict__ out2, int k) {
  int tile0 = blockIdx.x * 4;                 // grid = N2CV/64 = 1728 blocks
  int lane = threadIdx.x & 31;
  int nb   = threadIdx.x >> 5;
  int m    = lane & 15;
  int hi   = (lane >> 4) & 1;
  int n    = nb * 16 + (lane & 15);

  // B fragments: constant across all tiles for this (k, nb)
  const _Float16* bp = pw2 + ((((size_t)k * 8 + nb) * 2) * 32 + lane) * 16;
  v16h b0 = *(const v16h*)(bp);
  v16h b1 = *(const v16h*)(bp + 32 * 16);

  const int* imk0 = im + (size_t)k * N2CV + tile0 * 16;
  const int* omk0 = om + (size_t)k * N2CV + tile0 * 16;

#pragma unroll
  for (int tt = 0; tt < 4; tt += 2) {
    const int* imkA = imk0 + tt * 16;
    const int* imkB = imk0 + (tt + 1) * 16;
    int rowA = imkA[m];
    int rowB = imkB[m];
    const _Float16* arA = o1h + (size_t)rowA * 64 + hi * 8;
    const _Float16* arB = o1h + (size_t)rowB * 64 + hi * 8;
    v16h a0A = load_a_frag(arA, 0);
    v16h a0B = load_a_frag(arB, 0);
    v16h a1A = load_a_frag(arA, 1);
    v16h a1B = load_a_frag(arB, 1);
    v8f cA = {}, cB = {};
    cA = __builtin_amdgcn_wmma_f32_16x16x32_f16(false, a0A, false, b0, (short)0, cA, false, false);
    cB = __builtin_amdgcn_wmma_f32_16x16x32_f16(false, a0B, false, b0, (short)0, cB, false, false);
    cA = __builtin_amdgcn_wmma_f32_16x16x32_f16(false, a1A, false, b1, (short)0, cA, false, false);
    cB = __builtin_amdgcn_wmma_f32_16x16x32_f16(false, a1B, false, b1, (short)0, cB, false, false);
    scatter_add(out2, omk0 + tt * 16, hi, n, cA);
    scatter_add(out2, omk0 + (tt + 1) * 16, hi, n, cB);
  }
}

// ---- BN stats: deterministic two-stage reduction -------------------------
template <int C, int ROWS>
__global__ void __launch_bounds__(256) bn_stats(const float* __restrict__ x,
                                                const unsigned char* __restrict__ mask,
                                                int nrows, float* __restrict__ psum,
                                                float* __restrict__ psq, int* __restrict__ pcnt) {
  constexpr int G = 256 / C;
  __shared__ float ls[256], ls2[256];
  __shared__ int lc[G];
  int t = threadIdx.x, c = t % C, g = t / C;
  int base = blockIdx.x * ROWS;
  float s = 0.f, s2 = 0.f; int cn = 0;
  for (int r = base + g; r < base + ROWS && r < nrows; r += G) {
    if (mask[r]) {
      float v = x[(size_t)r * C + c];
      s += v; s2 += v * v; cn++;
    }
  }
  ls[t] = s; ls2[t] = s2;
  if (c == 0) lc[g] = cn;
  __syncthreads();
  if (g == 0) {
    for (int gg = 1; gg < G; ++gg) { s += ls[gg * C + c]; s2 += ls2[gg * C + c]; }
    psum[(size_t)blockIdx.x * C + c] = s;
    psq[(size_t)blockIdx.x * C + c] = s2;
    if (c == 0) {
      int tot = 0;
      for (int gg = 0; gg < G; ++gg) tot += lc[gg];
      pcnt[blockIdx.x] = tot;
    }
  }
}

template <int C>
__global__ void bn_reduce(const float* __restrict__ psum, const float* __restrict__ psq,
                          const int* __restrict__ pcnt, int nblocks,
                          float* __restrict__ sums, float* __restrict__ sumsq,
                          int* __restrict__ cnt) {
  int c = threadIdx.x;   // blockDim == C
  float s = 0.f, s2 = 0.f;
  for (int b = 0; b < nblocks; ++b) { s += psum[(size_t)b * C + c]; s2 += psq[(size_t)b * C + c]; }
  sums[c] = s; sumsq[c] = s2;
  if (c == 0) {
    int t = 0;
    for (int b = 0; b < nblocks; ++b) t += pcnt[b];
    cnt[0] = t;
  }
}

// ---- BN1 apply + ReLU + mask, emit f16 activations -----------------------
__global__ void __launch_bounds__(256) bn1_apply(const float* __restrict__ x,
                                                 const unsigned char* __restrict__ mask,
                                                 const float* __restrict__ g,
                                                 const float* __restrict__ b,
                                                 const float* __restrict__ sums,
                                                 const float* __restrict__ sumsq,
                                                 const int* __restrict__ cnt,
                                                 _Float16* __restrict__ y) {
  size_t idx = (size_t)blockIdx.x * 256 + threadIdx.x;   // N1CV*64
  int c = (int)(idx & 63);
  size_t r = idx >> 6;
  float n = fmaxf((float)cnt[0], 1.f);
  float mean = sums[c] / n;
  float var  = fmaxf(sumsq[c] / n - mean * mean, 0.f);
  float v = (x[idx] - mean) * rsqrtf(var + BN_EPS) * g[c] + b[c];
  v = fmaxf(v, 0.f);
  y[idx] = (_Float16)(mask[r] ? v : 0.f);
}

// ---- batch segment boundaries (batch_idx is sorted by construction) ------
__global__ void init_starts(int* start) { if (threadIdx.x < 9) start[threadIdx.x] = N2CV; }
__global__ void __launch_bounds__(256) find_starts(const int* __restrict__ bidx, int* start) {
  int i = blockIdx.x * 256 + threadIdx.x;
  if (i < N2CV) {
    int b = bidx[i];
    if (b >= 0 && b <= 8) atomicMin(&start[b], i);
  }
}

// ---- fused BN2 + ReLU + per-batch mean pool ------------------------------
__global__ void __launch_bounds__(256) pool_kernel(const float* __restrict__ out2,
                                                   const float* __restrict__ g2,
                                                   const float* __restrict__ b2,
                                                   const float* __restrict__ sums2,
                                                   const float* __restrict__ sumsq2,
                                                   const int* __restrict__ cnt2,
                                                   const int* __restrict__ start,
                                                   float* __restrict__ pooled) {
  __shared__ float ls[256];
  int t = threadIdx.x, c = t & 127, h = t >> 7;
  int b = blockIdx.x;
  float n = fmaxf((float)cnt2[0], 1.f);
  float mean  = sums2[c] / n;
  float var   = fmaxf(sumsq2[c] / n - mean * mean, 0.f);
  float scale = rsqrtf(var + BN_EPS) * g2[c];
  float shift = b2[c];
  int r0 = start[b], r1 = start[b + 1];
  float s = 0.f;
  for (int r = r0 + h; r < r1; r += 2) {
    float v = (out2[(size_t)r * 128 + c] - mean) * scale + shift;
    s += fmaxf(v, 0.f);
  }
  ls[t] = s;
  __syncthreads();
  if (h == 0) {
    float tot = s + ls[128 + c];
    pooled[b * 128 + c] = tot / fmaxf((float)(r1 - r0), 1.f);
  }
}

// ---- final linear 128 -> 40 ----------------------------------------------
__global__ void final_linear(const float* __restrict__ pooled, const float* __restrict__ Wl,
                             const float* __restrict__ bl, float* __restrict__ out) {
  int t = threadIdx.x;
  if (t >= 320) return;
  int b = t / 40, j = t % 40;
  float s = bl[j];
  const float* p = pooled + b * 128;
  const float* w = Wl + (size_t)j * 128;
  for (int c = 0; c < 128; ++c) s += p[c] * w[c];
  out[b * 40 + j] = s;
}

extern "C" void kernel_launch(void* const* d_in, const int* in_sizes, int n_in,
                              void* d_out, int out_size, void* d_ws, size_t ws_size,
                              hipStream_t stream) {
  (void)in_sizes; (void)n_in; (void)out_size; (void)ws_size;
  const float* feats = (const float*)d_in[0];
  const float* W1    = (const float*)d_in[1];
  const float* g1    = (const float*)d_in[2];
  const float* b1    = (const float*)d_in[3];
  const float* W2    = (const float*)d_in[4];
  const float* g2    = (const float*)d_in[5];
  const float* b2    = (const float*)d_in[6];
  const float* Wl    = (const float*)d_in[7];
  const float* bl    = (const float*)d_in[8];
  const int*   im1   = (const int*)d_in[9];
  const int*   om1   = (const int*)d_in[10];
  const unsigned char* mask1 = (const unsigned char*)d_in[11];
  const int*   im2   = (const int*)d_in[12];
  const int*   om2   = (const int*)d_in[13];
  const unsigned char* mask2 = (const unsigned char*)d_in[14];
  const int*   bidx  = (const int*)d_in[15];
  float* out = (float*)d_out;

  // ---- carve workspace (256B aligned chunks) ----
  char* wsb = (char*)d_ws;
  size_t off = 0;
  auto take = [&](size_t bytes) { char* p = wsb + off; off += (bytes + 255) & ~(size_t)255; return p; };
  const size_t sz_out1 = (size_t)(N1CV + 1) * 64 * 4;
  const size_t sz_out2 = (size_t)(N2CV + 1) * 128 * 4;
  float*    out1  = (float*)take(sz_out1);
  float*    out2  = (float*)take(sz_out2);
  _Float16* o1h   = (_Float16*)take((size_t)N1CV * 64 * 2);
  _Float16* pw2   = (_Float16*)take((size_t)221184 * 2);
  float*    stats = (float*)take(512 * 4);      // sums1|sumsq1|sums2|sumsq2|cnt1|cnt2
  float* sums1  = stats;        float* sumsq1 = stats + 64;
  float* sums2  = stats + 128;  float* sumsq2 = stats + 256;
  int*   cnt1   = (int*)(stats + 384);
  int*   cnt2   = cnt1 + 1;
  float* psum   = (float*)take((size_t)1024 * 128 * 4);
  float* psq    = (float*)take((size_t)1024 * 128 * 4);
  int*   pcnt   = (int*)take(1024 * 4);
  int*   starts = (int*)take(16 * 4);
  float* pooled = (float*)take(1024 * 4);

  // ---- zero accumulators ----
  hipMemsetAsync(out1, 0, sz_out1, stream);
  hipMemsetAsync(out2, 0, sz_out2, stream);

  // ---- pack W2 into WMMA B-fragment layout ----
  pack_w2_kernel<<<221184 / 256, 256, 0, stream>>>(W2, pw2);

  // ---- conv1: 27 race-free passes ----
  for (int k = 0; k < 27; ++k)
    conv1_pass<<<(N1CV * 4) / 256, 256, 0, stream>>>(feats, W1, im1, om1, out1, k);

  // ---- BN1 stats (deterministic) + apply ----
  const int n1blk = (N1CV + 511) / 512;   // 782
  bn_stats<64, 512><<<n1blk, 256, 0, stream>>>(out1, mask1, N1CV, psum, psq, pcnt);
  bn_reduce<64><<<1, 64, 0, stream>>>(psum, psq, pcnt, n1blk, sums1, sumsq1, cnt1);
  bn1_apply<<<(N1CV * 64) / 256, 256, 0, stream>>>(out1, mask1, g1, b1, sums1, sumsq1, cnt1, o1h);

  // ---- conv2: 27 WMMA passes (4 tiles per block) ----
  for (int k = 0; k < 27; ++k)
    conv2_wmma<<<N2CV / 64, 256, 0, stream>>>(o1h, pw2, im2, om2, out2, k);

  // ---- BN2 stats ----
  const int n2blk = (N2CV + 511) / 512;   // 216
  bn_stats<128, 512><<<n2blk, 256, 0, stream>>>(out2, mask2, N2CV, psum, psq, pcnt);
  bn_reduce<128><<<1, 128, 0, stream>>>(psum, psq, pcnt, n2blk, sums2, sumsq2, cnt2);

  // ---- segment boundaries + fused BN2/ReLU/mean-pool ----
  init_starts<<<1, 32, 0, stream>>>(starts);
  find_starts<<<(N2CV + 255) / 256, 256, 0, stream>>>(bidx, starts);
  pool_kernel<<<8, 256, 0, stream>>>(out2, g2, b2, sums2, sumsq2, cnt2, starts, pooled);

  // ---- final linear ----
  final_linear<<<1, 384, 0, stream>>>(pooled, Wl, bl, out);
}
